// ShapeGuidedDiffusionPredictorFinal_72945724555743
// MI455X (gfx1250) — matrix operand order repaired and checked by reading, not compiled, not run on hardware
//
#include <hip/hip_runtime.h>
#include <hip/hip_bf16.h>

typedef __attribute__((ext_vector_type(16))) __bf16 v16bf;
typedef __attribute__((ext_vector_type(8)))  float  v8f;

#define MID   64
#define KSQ   9
#define NOUT  640          // MID*(1+KSQ)
#define CIN   256
#define HW    16384        // 128*128
#define NBATCH 8
#define PIX_PER_WG 32
#define NT_PER_WAVE 10     // 640/16 tiles / 4 wave-groups

// ---------------- Kernel 0: W fp32 -> bf16 in workspace, zero score accum ----
__global__ void prep_kernel(const float* __restrict__ W,
                            __bf16* __restrict__ Wb,
                            float* __restrict__ scores) {
  int idx = blockIdx.x * 256 + threadIdx.x;
  if (idx < NOUT * CIN) Wb[idx] = (__bf16)W[idx];
  if (idx < NBATCH * MID) scores[idx] = 0.0f;
}

// ---------------- Kernel 1: fused bf16-WMMA GEMM + bias + softmax epilogue ---
__global__ __launch_bounds__(256)
void fused_gemm_softmax(const float* __restrict__ x,
                        const __bf16* __restrict__ Wb,
                        const float* __restrict__ bias,
                        float* __restrict__ out,
                        float* __restrict__ scores) {
  // raw tile for this WG's 32 pixels, all 640 output channels.
  // Row stride 641 dwords -> (641 mod 64)==1 -> conflict-free column walks.
  __shared__ float raw[PIX_PER_WG][NOUT + 1];   // 82,048 bytes

  const int t    = threadIdx.x;
  // wave index and derived values are wave-uniform: force into SGPRs so the
  // N-tile loop is a scalar branch (EXEC stays all-ones around WMMA).
  const int wave = __builtin_amdgcn_readfirstlane(t >> 5);
  const int lane = t & 31;
  const int half = lane >> 4;     // lane group 0-15 vs 16-31 (WMMA layout)
  const int m    = lane & 15;
  const int mw   = wave & 1;      // which 16-pixel M subtile
  const int nq   = wave >> 1;     // which quarter of the 40 N tiles

  const int bimg = blockIdx.x >> 9;                 // 512 WGs per image
  const int p0   = (blockIdx.x & 511) * PIX_PER_WG; // pixel base in image
  const int pix  = p0 + mw * 16 + m;                // this lane's pixel row (A matrix M)

  const float* xb = x + (size_t)bimg * CIN * HW + pix;

  // ---- Load A operand (16 pixels x K=256) as bf16, contiguous K layout:
  //  lanes 0-15:  elems 0-15 = K0..15
  //  lanes 16-31: elems 0-15 = K16..31
  v16bf a[8];
#pragma unroll
  for (int kb = 0; kb < 8; ++kb) {
    const int c0 = kb * 32 + half * 16;
#pragma unroll
    for (int i = 0; i < 8; ++i) {
      a[kb][i]     = (__bf16)xb[(size_t)(c0 + i) * HW];        // K = c0+i
      a[kb][8 + i] = (__bf16)xb[(size_t)(c0 + 8 + i) * HW];    // K = c0+8+i
    }
  }

  // ---- GEMM: 5 iterations x 2 N tiles, two independent WMMA accumulation
  //      chains interleaved to hide XDL latency. 16 v_wmma per iteration.
  const int ntbase = nq * NT_PER_WAVE;
#pragma unroll 1
  for (int nt2 = 0; nt2 < NT_PER_WAVE / 2; ++nt2) {
    const int n0 = (ntbase + 2 * nt2) * 16 + m;     // lane's output column, tile A
    const int n1 = n0 + 16;                         // tile B
    const __bf16* wrow0 = Wb + (size_t)n0 * CIN + half * 16;
    const __bf16* wrow1 = Wb + (size_t)n1 * CIN + half * 16;
    v8f acc0 = {};
    v8f acc1 = {};
#pragma unroll
    for (int kb = 0; kb < 8; ++kb) {
      // B layout (32x16): lane n=l%16 holds 16 contiguous K at half*16
      v16bf bf0, bf1;
#pragma unroll
      for (int i = 0; i < 16; ++i) {
        bf0[i] = wrow0[kb * 32 + i];
        bf1[i] = wrow1[kb * 32 + i];
      }
      acc0 = __builtin_amdgcn_wmma_f32_16x16x32_bf16(
          false, a[kb], false, bf0, (short)0, acc0, false, false);
      acc1 = __builtin_amdgcn_wmma_f32_16x16x32_bf16(
          false, a[kb], false, bf1, (short)0, acc1, false, false);
    }
    // C layout: VGPR r, lanes 0-15 -> M=r, lanes 16-31 -> M=r+8 ; N = l%16
    const float bn0 = bias[n0];
    const float bn1 = bias[n1];
#pragma unroll
    for (int r = 0; r < 8; ++r) {
      raw[mw * 16 + half * 8 + r][n0] = acc0[r] + bn0;
      raw[mw * 16 + half * 8 + r][n1] = acc1[r] + bn1;
    }
  }
  __syncthreads();

  // ---- Score-map partial sums (channels 0..63) -> global atomics ------------
  if (t < MID) {
    float s = 0.0f;
#pragma unroll 4
    for (int p = 0; p < PIX_PER_WG; ++p) s += raw[p][t];  // conflict-free
    atomicAdd(&scores[bimg * MID + t], s);
  }

  // ---- Softmax over 9 taps per (pixel, mid-channel), write norm_kernels -----
  // Output is write-once, never re-read: use non-temporal stores to keep
  // x / W resident in L2.
  float* normk = out + NBATCH * MID;   // mask occupies first 512 floats
#pragma unroll 1
  for (int it = 0; it < (PIX_PER_WG * MID) / 256; ++it) {   // 8 iterations
    const int gid = t + it * 256;
    const int p   = gid & (PIX_PER_WG - 1);   // lane-contiguous -> coalesced
    const int g   = gid >> 5;                 // mid channel 0..63
    float v[KSQ];
    float mx = -3.0e38f;
#pragma unroll
    for (int j = 0; j < KSQ; ++j) {
      v[j] = raw[p][MID + g * KSQ + j];
      mx = fmaxf(mx, v[j]);
    }
    float s = 0.0f;
#pragma unroll
    for (int j = 0; j < KSQ; ++j) { v[j] = __expf(v[j] - mx); s += v[j]; }
    const float inv = 1.0f / s;
    const size_t base = (size_t)(bimg * MID + g) * KSQ * HW + (size_t)(p0 + p);
#pragma unroll
    for (int j = 0; j < KSQ; ++j)
      __builtin_nontemporal_store(v[j] * inv, &normk[base + (size_t)j * HW]);
  }
}

// ---------------- Kernel 2: sigmoid means, k, top-k, mask, indices -----------
__global__ void finalize_kernel(const float* __restrict__ scores,
                                float* __restrict__ out) {
  __shared__ float sg[NBATCH * MID];
  __shared__ int   s_k;
  const int t = threadIdx.x;
  if (t < NBATCH * MID)
    sg[t] = 1.0f / (1.0f + __expf(-scores[t] * (1.0f / (float)HW)));
  __syncthreads();
  if (t == 0) {
    float s = 0.0f;
#pragma clang loop unroll(disable)
    for (int i = 0; i < NBATCH * MID; ++i) s += sg[i];
    float scale = s / (float)(NBATCH * MID);
    scale = fminf(fmaxf(scale, 0.25f), 1.0f);
    int k = (int)ceilf((float)MID * scale);
    int kmin = (MID / 4 > 4) ? MID / 4 : 4;
    if (k < kmin) k = kmin;
    if (k > MID)  k = MID;
    s_k = k;
  }
  __syncthreads();
  const int k = s_k;
  if (t < NBATCH) {
    const float* gs = &sg[t * MID];
    int  order[MID];
    bool taken[MID];
#pragma clang loop unroll(disable)
    for (int o = 0; o < MID; ++o) taken[o] = false;
#pragma clang loop unroll(disable)
    for (int r = 0; r < MID; ++r) {            // stable descending selection
      int best = 0; float bv = -3.0e38f;
#pragma clang loop unroll(disable)
      for (int o = 0; o < MID; ++o)
        if (!taken[o] && gs[o] > bv) { bv = gs[o]; best = o; }
      taken[best] = true; order[r] = best;
    }
#pragma clang loop unroll(disable)
    for (int o = 0; o < MID; ++o) out[t * MID + o] = 0.0f;
#pragma clang loop unroll(disable)
    for (int r = 0; r < k; ++r)  out[t * MID + order[r]] = 1.0f;
    const size_t idx_base =
        (size_t)NBATCH * MID + (size_t)NBATCH * MID * KSQ * HW + 1;
#pragma clang loop unroll(disable)
    for (int r = 0; r < k; ++r)
      out[idx_base + (size_t)t * k + r] = (float)order[r];
  }
  if (t == 0)
    out[(size_t)NBATCH * MID + (size_t)NBATCH * MID * KSQ * HW] = (float)s_k;
}

// ---------------- launch ------------------------------------------------------
extern "C" void kernel_launch(void* const* d_in, const int* in_sizes, int n_in,
                              void* d_out, int out_size, void* d_ws, size_t ws_size,
                              hipStream_t stream) {
  const float* x    = (const float*)d_in[0];
  const float* W    = (const float*)d_in[1];
  const float* bias = (const float*)d_in[2];
  float* out = (float*)d_out;

  __bf16* Wb     = (__bf16*)d_ws;                                  // 327,680 B
  float*  scores = (float*)((char*)d_ws + (size_t)NOUT * CIN * 2); // 2,048 B

  prep_kernel<<<(NOUT * CIN + 255) / 256, 256, 0, stream>>>(W, Wb, scores);

  const int wgs = NBATCH * (HW / PIX_PER_WG);   // 4096 workgroups
  fused_gemm_softmax<<<wgs, 256, 0, stream>>>(x, Wb, bias, out, scores);

  finalize_kernel<<<1, 512, 0, stream>>>(scores, out);
}
